// PatchedSparseNetwork_45518063403493
// MI455X (gfx1250) — matrix operand order repaired and checked by reading, not compiled, not run on hardware
//
#include <hip/hip_runtime.h>
#include <cstdint>

typedef unsigned short u16;
typedef __bf16 bf16;
typedef __attribute__((ext_vector_type(16))) __bf16 bf16x16;
typedef __attribute__((ext_vector_type(8)))  float  f32x8;

union Frag {
    bf16x16 v;
    uint4   q[2];
};

__device__ __forceinline__ u16 f2bf(float f) {
    unsigned int u = __float_as_uint(f);
    u += 0x7FFFu + ((u >> 16) & 1u);   // round-to-nearest-even
    return (u16)(u >> 16);
}

// ---- CDNA5 async global->LDS copy (tracked by ASYNCcnt) ---------------------
__device__ __forceinline__ void async_copy_b128(uint32_t lds_off, const void* gptr) {
    asm volatile("global_load_async_to_lds_b128 %0, %1, off"
                 :: "v"(lds_off), "v"(gptr)
                 : "memory");
}

__device__ __forceinline__ void wait_async0() {
#if __has_builtin(__builtin_amdgcn_s_wait_asynccnt)
    __builtin_amdgcn_s_wait_asynccnt(0);
#else
    asm volatile("s_wait_asynccnt 0x0" ::: "memory");
#endif
}

// ---- elementwise prep: Wm = bf16(W * m) -------------------------------------
__global__ void mask_cast_kernel(const float* __restrict__ W,
                                 const float* __restrict__ m,
                                 u16* __restrict__ out) {
    size_t j = ((size_t)blockIdx.x * blockDim.x + threadIdx.x) * 4;
    float4 w  = *(const float4*)(W + j);
    float4 mm = *(const float4*)(m + j);
    unsigned int lo = (unsigned)f2bf(w.x * mm.x) | ((unsigned)f2bf(w.y * mm.y) << 16);
    unsigned int hi = (unsigned)f2bf(w.z * mm.z) | ((unsigned)f2bf(w.w * mm.w) << 16);
    *(uint2*)(out + j) = uint2{lo, hi};
}

__global__ void cast_kernel(const float* __restrict__ X, u16* __restrict__ out) {
    size_t j = ((size_t)blockIdx.x * blockDim.x + threadIdx.x) * 4;
    float4 w = *(const float4*)(X + j);
    unsigned int lo = (unsigned)f2bf(w.x) | ((unsigned)f2bf(w.y) << 16);
    unsigned int hi = (unsigned)f2bf(w.z) | ((unsigned)f2bf(w.w) << 16);
    *(uint2*)(out + j) = uint2{lo, hi};
}

// ---- epilogue store helpers -------------------------------------------------
__device__ __forceinline__ void store_out(float* p, float v) { *p = v; }
__device__ __forceinline__ void store_out(u16* p, float v)   { *p = f2bf(v); }

// ---- GEMM: C[M,N] = act(A[M,K] * W[N,K]^T + bias) ---------------------------
// A, W are bf16 row-major with K contiguous; accumulate f32 via WMMA.
// Block: 256 threads = 8 waves. Tile: 128(M) x 128(N), BK=64.
// LDS double-buffered, panels fetched with global_load_async_to_lds_b128.
// Wave grid 2(M) x 4(N): each wave computes 64x32 = 4x2 tiles of 16x16.
template <bool RELU, typename OutT>
__global__ __launch_bounds__(256, 1)
void gemm_bf16_wmma(const u16* __restrict__ A, const u16* __restrict__ W,
                    const float* __restrict__ bias, OutT* __restrict__ C,
                    int M, int N, int K) {
    constexpr int BM = 128, BK = 64, LDT = 72;  // 72*2B = 144B row stride (16B multiple)
    __shared__ u16 As[2][BM * LDT];
    __shared__ u16 Bs[2][BM * LDT];

    const int tid = threadIdx.x;
    const int bm = blockIdx.y, bn = blockIdx.x;

    // loader mapping: each thread moves 4x16B for A and 4x16B for B per panel
    const int lrow = tid >> 3;          // 0..31
    const int lcol = (tid & 7) * 8;     // bf16 elems, 16B chunks
    const size_t aBase = (size_t)(bm * BM + lrow) * K + lcol;
    const size_t bBase = (size_t)(bn * BM + lrow) * K + lcol;
    const uint32_t ldsA = (uint32_t)(uintptr_t)(&As[0][lrow * LDT + lcol]);
    const uint32_t ldsB = (uint32_t)(uintptr_t)(&Bs[0][lrow * LDT + lcol]);
    constexpr uint32_t bufStride = (uint32_t)(BM * LDT) * 2u;   // bytes per buffer

    // issue panel 0 into buffer 0
    #pragma unroll
    for (int c = 0; c < 4; c++) {
        async_copy_b128(ldsA + (uint32_t)(32 * c * LDT) * 2u,
                        A + aBase + (size_t)(32 * c) * K);
        async_copy_b128(ldsB + (uint32_t)(32 * c * LDT) * 2u,
                        W + bBase + (size_t)(32 * c) * K);
    }
    wait_async0();
    __syncthreads();

    const int wave = tid >> 5, lane = tid & 31;
    const int wm = wave >> 2, wn = wave & 3;     // 2x4 wave grid
    const int half = lane >> 4, lr = lane & 15;

    f32x8 acc[4][2];
    #pragma unroll
    for (int i = 0; i < 4; i++)
        #pragma unroll
        for (int j = 0; j < 2; j++)
            #pragma unroll
            for (int e = 0; e < 8; e++) acc[i][j][e] = 0.0f;

    const int panels = K / BK;
    for (int p = 0; p < panels; p++) {
        const int buf = p & 1;
        if (p + 1 < panels) {
            // fetch next panel into the other buffer (everyone finished reading
            // it before the previous barrier), overlapping with WMMA below
            const uint32_t nb = (uint32_t)(buf ^ 1) * bufStride;
            const size_t koff = (size_t)(p + 1) * BK;
            #pragma unroll
            for (int c = 0; c < 4; c++) {
                async_copy_b128(ldsA + nb + (uint32_t)(32 * c * LDT) * 2u,
                                A + aBase + (size_t)(32 * c) * K + koff);
                async_copy_b128(ldsB + nb + (uint32_t)(32 * c * LDT) * 2u,
                                W + bBase + (size_t)(32 * c) * K + koff);
            }
        }
        #pragma unroll
        for (int kk = 0; kk < BK; kk += 32) {
            Frag a[4], b[2];
            // A tile 16x32: lanes 0-15 hold K {0..7,16..23}, lanes 16-31 hold K {8..15,24..31}
            #pragma unroll
            for (int mt = 0; mt < 4; mt++) {
                const u16* s = &As[buf][(wm * 64 + mt * 16 + lr) * LDT + kk + half * 8];
                a[mt].q[0] = *(const uint4*)(s);
                a[mt].q[1] = *(const uint4*)(s + 16);
            }
            // B tile 32x16: lanes 0-15 hold K {0..7,16..23}, lanes 16-31 hold K {8..15,24..31} (lane = N col)
            #pragma unroll
            for (int nt = 0; nt < 2; nt++) {
                const u16* s = &Bs[buf][(wn * 32 + nt * 16 + lr) * LDT + kk + half * 8];
                b[nt].q[0] = *(const uint4*)(s);
                b[nt].q[1] = *(const uint4*)(s + 16);
            }
            #pragma unroll
            for (int mt = 0; mt < 4; mt++)
                #pragma unroll
                for (int nt = 0; nt < 2; nt++)
                    acc[mt][nt] = __builtin_amdgcn_wmma_f32_16x16x32_bf16(
                        false, a[mt].v, false, b[nt].v, (short)0, acc[mt][nt],
                        false, false);
        }
        if (p + 1 < panels) {
            wait_async0();
            __syncthreads();
        }
    }

    // C 16x16 f32 layout: lane = N col (mod 16); VGPR v -> M = half*8 + v
    const int cm0 = bm * BM + wm * 64;
    const int cn0 = bn * BM + wn * 32;
    #pragma unroll
    for (int nt = 0; nt < 2; nt++) {
        const int n = cn0 + nt * 16 + lr;
        const float bv = bias[n];
        #pragma unroll
        for (int mt = 0; mt < 4; mt++) {
            #pragma unroll
            for (int v = 0; v < 8; v++) {
                const int mrow = cm0 + mt * 16 + half * 8 + v;
                float val = acc[mt][nt][v] + bv;
                if (RELU) val = val > 0.0f ? val : 0.0f;
                store_out(&C[(size_t)mrow * N + n], val);
            }
        }
    }
}

// ---- launch -----------------------------------------------------------------
extern "C" void kernel_launch(void* const* d_in, const int* in_sizes, int n_in,
                              void* d_out, int out_size, void* d_ws, size_t ws_size,
                              hipStream_t stream) {
    const float* x  = (const float*)d_in[0];
    const float* W0 = (const float*)d_in[1];
    const float* b0 = (const float*)d_in[2];
    const float* m0 = (const float*)d_in[3];
    const float* W1 = (const float*)d_in[4];
    const float* b1 = (const float*)d_in[5];
    const float* m1 = (const float*)d_in[6];
    const float* W2 = (const float*)d_in[7];
    const float* b2 = (const float*)d_in[8];
    const float* m2 = (const float*)d_in[9];
    const float* W3 = (const float*)d_in[10];
    const float* b3 = (const float*)d_in[11];
    const float* m3 = (const float*)d_in[12];
    float* out = (float*)d_out;

    const int B = 8192;
    const size_t nX  = (size_t)B * 4096;
    const size_t nH1 = (size_t)B * 8192;
    const size_t nW0 = (size_t)8192 * 4096;
    const size_t nW1 = (size_t)8192 * 8192;
    const size_t nW2 = (size_t)4096 * 8192;
    const size_t nW3 = (size_t)1024 * 4096;

    char* ws = (char*)d_ws;
    u16* xb  = (u16*)ws;  ws += nX  * 2;
    u16* h1  = (u16*)ws;  ws += nH1 * 2;
    u16* h2  = (u16*)ws;  ws += nH1 * 2;
    u16* h3  = (u16*)ws;  ws += nX  * 2;
    u16* Wm0 = (u16*)ws;  ws += nW0 * 2;
    u16* Wm1 = (u16*)ws;  ws += nW1 * 2;
    u16* Wm2 = (u16*)ws;  ws += nW2 * 2;
    u16* Wm3 = (u16*)ws;  ws += nW3 * 2;

    // prep: bf16 masked weights + bf16 input
    cast_kernel<<<(unsigned)(nX / 1024), 256, 0, stream>>>(x, xb);
    mask_cast_kernel<<<(unsigned)(nW0 / 1024), 256, 0, stream>>>(W0, m0, Wm0);
    mask_cast_kernel<<<(unsigned)(nW1 / 1024), 256, 0, stream>>>(W1, m1, Wm1);
    mask_cast_kernel<<<(unsigned)(nW2 / 1024), 256, 0, stream>>>(W2, m2, Wm2);
    mask_cast_kernel<<<(unsigned)(nW3 / 1024), 256, 0, stream>>>(W3, m3, Wm3);

    dim3 blk(256);
    gemm_bf16_wmma<true,  u16><<<dim3(8192 / 128, B / 128), blk, 0, stream>>>(
        xb, Wm0, b0, h1, B, 8192, 4096);
    gemm_bf16_wmma<true,  u16><<<dim3(8192 / 128, B / 128), blk, 0, stream>>>(
        h1, Wm1, b1, h2, B, 8192, 8192);
    gemm_bf16_wmma<true,  u16><<<dim3(4096 / 128, B / 128), blk, 0, stream>>>(
        h2, Wm2, b2, h3, B, 4096, 8192);
    gemm_bf16_wmma<false, float><<<dim3(1024 / 128, B / 128), blk, 0, stream>>>(
        h3, Wm3, b3, out, B, 1024, 4096);
}
